// MultiHeadVoting_44203803410505
// MI455X (gfx1250) — matrix operand order repaired and checked by reading, not compiled, not run on hardware
//
#include <hip/hip_runtime.h>
#include <hip/hip_bf16.h>

typedef __attribute__((ext_vector_type(16))) _Float16 v16h;
typedef __attribute__((ext_vector_type(8)))  float    v8f;

namespace {
constexpr int kB       = 32;
constexpr int kHeads   = 12;
constexpr int kPatch   = 784;   // 28*28
constexpr int kSide    = 28;
constexpr int kRow     = 785;   // (PATCH_NUM + 1)
constexpr int kVotes   = 24;
constexpr int kThreads = 384;   // 12 waves of 32
} // namespace

// Wave32 argmax reduction; ties broken by lowest index (matches jax top_k /
// stable argsort of the negated array).
__device__ __forceinline__ void wave_argmax(float& bv, int& bi) {
#pragma unroll
  for (int off = 16; off; off >>= 1) {
    float ov = __shfl_xor(bv, off, 32);
    int   oi = __shfl_xor(bi, off, 32);
    if (ov > bv || (ov == bv && oi < bi)) { bv = ov; bi = oi; }
  }
}

// One 16x16 (M,N) output tile of a 32x32x32 f16 matmul, K=32 in a single
// v_wmma_f32_16x16x32_f16.
//   A: row-major 32-column LDS array (rows rbase..rbase+15 used).
//   Bt: COLUMN-major LDS array (Bt[n*32 + k] = B[k][n]) so each lane's B
//       fragment (K = kbase..kbase+15 of column cbase+n) is one contiguous,
//       32-byte-aligned v16h -> two ds_load_b128.
// Lane layouts used here (contiguous K order, same in A and B):
//   A (16x32 f16): M = lane&15; halves 0..7 -> K = akb+q, 8..15 -> K = akb+8+q,
//                  akb = (lane<16) ? 0 : 16.
//   B (32x16 f16): N = lane&15; half q -> K = bkb+q, bkb = (lane<16) ? 0 : 16.
__device__ __forceinline__ v8f wmma_tile_k32(const _Float16* __restrict__ A,
                                             const _Float16* __restrict__ Bt,
                                             int rbase, int cbase, int lane) {
  const int m   = lane & 15;
  const int akb = (lane & 16) ? 16 : 0;
  const _Float16* arow = A + (rbase + m) * 32;
  v16h a;
#pragma unroll
  for (int q = 0; q < 8; ++q) {
    a[q]     = arow[akb + q];        // 16B chunk -> ds_load_b128
    a[q + 8] = arow[akb + 8 + q];    // 16B chunk -> ds_load_b128
  }
  const int n   = lane & 15;
  const int bkb = (lane & 16) ? 16 : 0;
  const v16h b =
      *reinterpret_cast<const v16h*>(Bt + (cbase + n) * 32 + bkb);

  v8f c = {};
  c = __builtin_amdgcn_wmma_f32_16x16x32_f16(
      /*neg_a=*/false, a, /*neg_b=*/false, b,
      /*c_mod=*/(short)0, c, /*reuse_a=*/false, /*reuse_b=*/false);
  return c;
}

__global__ __launch_bounds__(kThreads)
void mhv_fused_kernel(const float* __restrict__ x,
                      const float* __restrict__ k3,
                      const int*   __restrict__ seln_p,
                      int*         __restrict__ out) {
  __shared__ float cnt[kPatch];                 // vote counts, 28x28 flattened
  __shared__ alignas(32) _Float16 Uh[1024];     // left factor, ROW-major (A op)
  __shared__ alignas(32) _Float16 Gt[1024];     // counts f16, COL-major (B op)
  __shared__ alignas(32) _Float16 Vt[1024];     // right factor, COL-major (B op)
  __shared__ alignas(32) _Float16 Hh[1024];     // H = U@G, ROW-major (A op)
  __shared__ float Of[1024];                    // O = H@V, final smoothed counts

  const int b    = (int)blockIdx.x;
  const int tid  = (int)threadIdx.x;
  const int lane = tid & 31;
  const int wave = tid >> 5;

  for (int i = tid; i < kPatch; i += kThreads) cnt[i] = 0.0f;
  __syncthreads();

  // ---- Phase 1: per-head top-24 voting (wave w owns head w) ----
  {
    const float* score =
        x + (size_t)(b * kHeads + wave) * kRow * kRow + 1;  // x[b, h, 0, 1:]
    float val[25];
#pragma unroll
    for (int j = 0; j < 25; ++j) {
      const int idx = j * 32 + lane;
      val[j] = (idx < kPatch) ? score[idx] : -__builtin_inff();
    }
    for (int it = 0; it < kVotes; ++it) {
      float bv = -__builtin_inff();
      int   bi = 0x7fffffff;
#pragma unroll
      for (int j = 0; j < 25; ++j) {
        if (val[j] > bv) { bv = val[j]; bi = j * 32 + lane; }
      }
      wave_argmax(bv, bi);
      // Winner's owning lane retires it (register select, no scratch spill).
      const bool mine = ((bi & 31) == lane);
      const int  kj   = bi >> 5;
#pragma unroll
      for (int j = 0; j < 25; ++j)
        if (mine && j == kj) val[j] = -__builtin_inff();
      if (lane == 0) atomicAdd(&cnt[bi], 1.0f);  // ds_add_f32
    }
  }
  __syncthreads();

  // ---- Phase 2: build factored-conv operands ----
  // kernel = outer(u, v) (rank-1): u_i = K[i][1], v_j = K[1][j] / K[1][1].
  // SAME conv == U @ G @ V with banded U[r][c] = u[c-r+1], V[r][c] = v[r-c+1];
  // zero padding of G to 32x32 reproduces the SAME boundary exactly.
  // G and V are stored transposed (column-major) for the WMMA B-operand path.
  {
    const float kc  = k3[4];
    const float u0  = k3[1], u1 = k3[4], u2 = k3[7];
    const float vv0 = k3[3] / kc, vv1 = 1.0f, vv2 = k3[5] / kc;
    for (int i = tid; i < 1024; i += kThreads) {
      // Row-major build of U: i = r*32 + c.
      const int r = i >> 5, c = i & 31;
      const int d = c - r;
      Uh[i] = (_Float16)((d == -1) ? u0 : (d == 0) ? u1 : (d == 1) ? u2 : 0.0f);
      // Column-major build of G and V: i = col*32 + row.
      const int col = i >> 5, row = i & 31;
      Gt[i] = (_Float16)((row < kSide && col < kSide) ? cnt[row * kSide + col]
                                                      : 0.0f);
      const int d2 = row - col;
      Vt[i] = (_Float16)((d2 == -1) ? vv0 : (d2 == 0) ? vv1
                                          : (d2 == 1) ? vv2 : 0.0f);
    }
  }
  __syncthreads();

  // ---- Phase 3: conv as two 32x32x32 WMMA matmuls (waves 0..3, EXEC full) ----
  if (wave < 4) {
    const int rbase = (wave >> 1) * 16;
    const int cbase = (wave & 1) * 16;
    const v8f acc   = wmma_tile_k32(Uh, Gt, rbase, cbase, lane);
    // D layout: VGPR q -> M = q + 8*(lane>=16), N = lane&15.
    const int mb = (lane & 16) ? 8 : 0;
    const int nc = lane & 15;
#pragma unroll
    for (int q = 0; q < 8; ++q)
      Hh[(rbase + mb + q) * 32 + cbase + nc] = (_Float16)acc[q];
  }
  __syncthreads();
  if (wave < 4) {
    const int rbase = (wave >> 1) * 16;
    const int cbase = (wave & 1) * 16;
    const v8f acc   = wmma_tile_k32(Hh, Vt, rbase, cbase, lane);
    const int mb = (lane & 16) ? 8 : 0;
    const int nc = lane & 15;
#pragma unroll
    for (int q = 0; q < 8; ++q)
      Of[(rbase + mb + q) * 32 + cbase + nc] = acc[q];
  }
  __syncthreads();

  // ---- Phase 4: select top-select_num smoothed bins, emit 1-based indices ----
  if (wave == 0) {
    int sn = seln_p[0];
    if (sn > kPatch) sn = kPatch;
    float val[25];
#pragma unroll
    for (int j = 0; j < 25; ++j) {
      const int idx = j * 32 + lane;
      val[j] = (idx < kPatch) ? Of[(idx / kSide) * 32 + (idx % kSide)]
                              : -__builtin_inff();
    }
    for (int it = 0; it < sn; ++it) {
      float bv = -__builtin_inff();
      int   bi = 0x7fffffff;
#pragma unroll
      for (int j = 0; j < 25; ++j) {
        if (val[j] > bv) { bv = val[j]; bi = j * 32 + lane; }
      }
      wave_argmax(bv, bi);
      const bool mine = ((bi & 31) == lane);
      const int  kj   = bi >> 5;
#pragma unroll
      for (int j = 0; j < 25; ++j)
        if (mine && j == kj) val[j] = -__builtin_inff();
      if (lane == 0) out[b * sn + it] = bi + 1;
    }
  }
}

extern "C" void kernel_launch(void* const* d_in, const int* in_sizes, int n_in,
                              void* d_out, int out_size, void* d_ws, size_t ws_size,
                              hipStream_t stream) {
  (void)in_sizes; (void)n_in; (void)out_size; (void)d_ws; (void)ws_size;
  const float* x    = (const float*)d_in[0];
  const float* k3   = (const float*)d_in[1];
  const int*   seln = (const int*)d_in[2];
  int*         out  = (int*)d_out;
  mhv_fused_kernel<<<dim3(kB), dim3(kThreads), 0, stream>>>(x, k3, seln, out);
}
